// LSTM_60069412602580
// MI455X (gfx1250) — matrix operand order repaired and checked by reading, not compiled, not run on hardware
//
#include <hip/hip_runtime.h>
#include <math.h>

typedef __attribute__((ext_vector_type(16))) _Float16 v16h;
typedef __attribute__((ext_vector_type(8)))  _Float16 v8h;
typedef __attribute__((ext_vector_type(8)))  float    v8f;

#define T_STEPS 1024
#define D_DIM   512
#define H_DIM   512
#define G_DIM   2048

// ---- LDS layout (byte offsets inside dynamic shared) ----
#define WB_OFF   0u                       // weight fragments: 2 mats x 8 waves x 16 kc x [2][32][16B] = 256 KB
#define WB_SIZE  262144u
#define XS_OFF   (WB_OFF + WB_SIZE)      // x stage, double buffered: 2 x 16 rows x 520 halves
#define XS_SIZE  16640u
#define HS_OFF   (XS_OFF + 2u * XS_SIZE) // h stage: 16 rows x 520 halves
#define A_OFF    (HS_OFF + XS_SIZE)      // gate accum: 4 x 16 x 36 f32
#define A_SIZE   9216u
#define C_OFF    (A_OFF + A_SIZE)        // cell state: 16 x 36 f32
#define C_SIZE   2304u
#define SMEM_TOTAL (C_OFF + C_SIZE)      // 323584 B  (<= 320 KB WGP LDS)
#define ROWP 520u                         // padded stage row stride (halves)

// ---------------- setup kernels ----------------

__global__ void lstm_pack_weights(const float* __restrict__ Wx,
                                  const float* __restrict__ Wh,
                                  _Float16* __restrict__ pWx,
                                  _Float16* __restrict__ pWh) {
    unsigned id = blockIdx.x * 256u + threadIdx.x;      // 0 .. 2M-1
    const float* W = Wx; _Float16* P = pWx;
    unsigned i = id;
    if (id >= (1u << 20)) { W = Wh; P = pWh; i = id - (1u << 20); }
    unsigned j8   = i & 7u;
    unsigned lane = (i >> 3) & 31u;
    unsigned h    = (i >> 8) & 1u;
    unsigned kc   = (i >> 9) & 15u;
    unsigned tile = i >> 13;                 // 0..127
    unsigned m  = lane & 15u;
    unsigned hi = lane >> 4;
    unsigned row = kc * 32u + h * 16u + (hi * 8u + j8);   // 0..511
    unsigned col = tile * 16u + m;                        // 0..2047
    P[i] = (_Float16)W[(size_t)row * G_DIM + col];
}

__global__ void lstm_convert_x(const float* __restrict__ x, _Float16* __restrict__ xh) {
    size_t id = (size_t)blockIdx.x * 256u + threadIdx.x;
    xh[id] = (_Float16)x[id];
}

__global__ void lstm_init_h(const float* __restrict__ h0, _Float16* __restrict__ hbuf) {
    unsigned id = blockIdx.x * 256u + threadIdx.x;    // 0..32767
    unsigned n = id >> 9, k = id & 511u;
    unsigned group = n >> 4, m = n & 15u;
    hbuf[32768u + group * 8192u + m * 512u + k] = (_Float16)h0[id];
}

// ---------------- device helpers ----------------

__device__ __forceinline__ void lstm_async_b128(unsigned lds_off, const void* gptr) {
    asm volatile("global_load_async_to_lds_b128 %0, %1, off"
                 :: "v"(lds_off), "v"(gptr) : "memory");
}
__device__ __forceinline__ void lstm_wait_async0() {
    asm volatile("s_wait_asynccnt 0x0" ::: "memory");
}
__device__ __forceinline__ v16h lstm_frag2(const _Float16* p0, const _Float16* p1) {
    v8h a = *(const v8h*)p0;
    v8h b = *(const v8h*)p1;
    return __builtin_shufflevector(a, b, 0,1,2,3,4,5,6,7,8,9,10,11,12,13,14,15);
}
__device__ __forceinline__ float lstm_sigmoid(float x) { return 1.0f / (1.0f + __expf(-x)); }
__device__ __forceinline__ float lstm_tanh(float x) {
    float e = __expf(-2.0f * fabsf(x));
    float r = (1.0f - e) / (1.0f + e);
    return copysignf(r, x);
}

// ---------------- persistent recurrent kernel ----------------

__global__ __launch_bounds__(256, 1)
void lstm_persistent(const _Float16* __restrict__ xh,
                     const _Float16* __restrict__ pWx,
                     const _Float16* __restrict__ pWh,
                     const float*    __restrict__ b,
                     _Float16*       __restrict__ hbuf,
                     int*            __restrict__ counters,
                     float*          __restrict__ out) {
    extern __shared__ __align__(16) unsigned char smem[];
    const unsigned smem_base = (unsigned)(size_t)smem;      // LDS byte offset
    const _Float16* sm = (const _Float16*)smem;             // half-indexed view
    float* aL = (float*)(smem + A_OFF);
    float* cL = (float*)(smem + C_OFF);

    const unsigned wg    = blockIdx.x;    // 0..63
    const unsigned group = wg >> 4;       // batch rows group*16..+15
    const unsigned w     = wg & 15u;      // h columns [w*32, w*32+32)
    const unsigned tid   = threadIdx.x;
    const unsigned lane  = tid & 31u;
    const unsigned wave  = tid >> 5;      // 0..7
    const unsigned g     = wave & 3u;     // gate (i,f,o,g)
    const unsigned s     = wave >> 2;     // sub-tile
    const unsigned m     = lane & 15u;
    const unsigned hi    = lane >> 4;
    const unsigned cb    = g * 512u + w * 32u + s * 16u;    // gate-column base
    const unsigned n0    = group * 16u;
    int* ctr = counters + group;

    // per-thread async-copy chunk coordinates (16 KB tile = 1024 x 16B chunks)
    unsigned crow[4], ccc[4];
    #pragma unroll
    for (int k = 0; k < 4; ++k) {
        unsigned c = tid + ((unsigned)k << 8);
        crow[k] = c >> 6; ccc[k] = c & 63u;
    }

    // ---- one-time: async-fill weight fragments into LDS (256 KB) ----
    for (unsigned k = 0; k < 64; ++k) {
        unsigned q = tid + (k << 8);
        unsigned seg = q >> 10, c = q & 1023u;
        unsigned mat = seg >> 3, wv = seg & 7u;
        unsigned tl = (wv & 3u) * 32u + w * 2u + (wv >> 2);       // tile of wave wv
        const _Float16* src = (mat ? pWh : pWx) + (size_t)tl * 8192u + c * 8u;
        lstm_async_b128(smem_base + WB_OFF + seg * 16384u + c * 16u, src);
    }
    // ---- one-time: stage x_0 into buffer 0 ----
    #pragma unroll
    for (int k = 0; k < 4; ++k) {
        const _Float16* src = xh + ((size_t)(n0 + crow[k]) * T_STEPS + 0) * D_DIM + ccc[k] * 8u;
        lstm_async_b128(smem_base + XS_OFF + crow[k] * 1040u + ccc[k] * 16u, src);
    }
    // zero cell state
    for (unsigned e = tid; e < 512u; e += 256u) cL[(e >> 5) * 36u + (e & 31u)] = 0.0f;

    const float bv = b[cb + m];
    const unsigned e0 = tid, e1 = tid + 256u;
    const unsigned row0 = e0 >> 5, col0 = e0 & 31u;
    const unsigned row1 = e1 >> 5, col1 = e1 & 31u;

    // fixed LDS base pointers (halves); per-chunk offsets become DS immediates
    const _Float16* pha = sm + (HS_OFF >> 1) + m * ROWP + hi * 8u;
    const _Float16* pbx = sm + (WB_OFF >> 1) + (wave * 16u) * 512u + lane * 8u;
    const _Float16* pbh = sm + (WB_OFF >> 1) + ((8u + wave) * 16u) * 512u + lane * 8u;

    for (int t = 0; t < T_STEPS; ++t) {
        const unsigned rb = (unsigned)(t + 1) & 1u;   // read h buffer
        const unsigned wb = (unsigned)t & 1u;         // write h buffer

        // stage h_{t-1}: global -> LDS (async)
        #pragma unroll
        for (int k = 0; k < 4; ++k) {
            const _Float16* src = hbuf + rb * 32768u + group * 8192u + crow[k] * 512u + ccc[k] * 8u;
            lstm_async_b128(smem_base + HS_OFF + crow[k] * 1040u + ccc[k] * 16u, src);
        }
        lstm_wait_async0();
        __syncthreads();

        // prefetch x_{t+1} into the other stage buffer (overlaps GEMM + barrier)
        if (t + 1 < T_STEPS) {
            #pragma unroll
            for (int k = 0; k < 4; ++k) {
                const _Float16* src = xh + ((size_t)(n0 + crow[k]) * T_STEPS + (t + 1)) * D_DIM + ccc[k] * 8u;
                lstm_async_b128(smem_base + XS_OFF + ((unsigned)(t + 1) & 1u) * XS_SIZE
                                + crow[k] * 1040u + ccc[k] * 16u, src);
            }
        }

        // ---- GEMM: a = b + x_t @ Wx + h @ Wh, fully unrolled, depth-3 pipeline ----
        const _Float16* pxa = sm + (XS_OFF >> 1) + ((unsigned)t & 1u) * (XS_SIZE >> 1)
                              + m * ROWP + hi * 8u;
        v8f acc = {bv, bv, bv, bv, bv, bv, bv, bv};

        #define LSTM_LA(u) ((u) < 16 ? lstm_frag2(pxa + (u) * 32u, pxa + (u) * 32u + 16u) \
                                     : lstm_frag2(pha + ((u) - 16) * 32u, pha + ((u) - 16) * 32u + 16u))
        #define LSTM_LB(u) ((u) < 16 ? lstm_frag2(pbx + (u) * 512u, pbx + (u) * 512u + 256u) \
                                     : lstm_frag2(pbh + ((u) - 16) * 512u, pbh + ((u) - 16) * 512u + 256u))
        v16h Af[3], Bf[3];
        #pragma unroll
        for (int u = 0; u < 3; ++u) { Af[u] = LSTM_LA(u); Bf[u] = LSTM_LB(u); }
        #pragma unroll
        for (int u = 0; u < 32; ++u) {
            v16h a = Af[u % 3], bb = Bf[u % 3];
            if (u + 3 < 32) { Af[u % 3] = LSTM_LA(u + 3); Bf[u % 3] = LSTM_LB(u + 3); }
            acc = __builtin_amdgcn_wmma_f32_16x16x32_f16(false, a, false, bb,
                                                         (short)0, acc, false, false);
        }
        #undef LSTM_LA
        #undef LSTM_LB

        #pragma unroll
        for (unsigned r = 0; r < 8; ++r)      // D[r][lane] = a[r+8*hi][cb+m]
            aL[g * 576u + (r + 8u * hi) * 36u + s * 16u + m] = acc[r];
        __syncthreads();

        // ---- gate mixing: 512 elements, 2 per thread ----
        {
            float ai = aL[row0 * 36u + col0],            af = aL[576u + row0 * 36u + col0];
            float ao = aL[1152u + row0 * 36u + col0],    ag = aL[1728u + row0 * 36u + col0];
            float c = lstm_sigmoid(af) * cL[row0 * 36u + col0] + lstm_sigmoid(ai) * lstm_tanh(ag);
            cL[row0 * 36u + col0] = c;
            float h = lstm_sigmoid(ao) * lstm_tanh(c);
            out[((size_t)(n0 + row0) * T_STEPS + t) * H_DIM + (w * 32u + col0)] = h;
            hbuf[wb * 32768u + group * 8192u + row0 * 512u + w * 32u + col0] = (_Float16)h;
        }
        {
            float ai = aL[row1 * 36u + col1],            af = aL[576u + row1 * 36u + col1];
            float ao = aL[1152u + row1 * 36u + col1],    ag = aL[1728u + row1 * 36u + col1];
            float c = lstm_sigmoid(af) * cL[row1 * 36u + col1] + lstm_sigmoid(ai) * lstm_tanh(ag);
            cL[row1 * 36u + col1] = c;
            float h = lstm_sigmoid(ao) * lstm_tanh(c);
            out[((size_t)(n0 + row1) * T_STEPS + t) * H_DIM + (w * 32u + col1)] = h;
            hbuf[wb * 32768u + group * 8192u + row1 * 512u + w * 32u + col1] = (_Float16)h;
        }
        __threadfence();        // release h_t to the other 15 WGs (L2)
        __syncthreads();

        // ---- inter-WG barrier over the 16 WGs of this batch group ----
        if (tid == 0) {
            __hip_atomic_fetch_add(ctr, 1, __ATOMIC_RELEASE, __HIP_MEMORY_SCOPE_AGENT);
            const int target = 16 * (t + 1);
            while (__hip_atomic_load(ctr, __ATOMIC_ACQUIRE, __HIP_MEMORY_SCOPE_AGENT) < target)
                __builtin_amdgcn_s_sleep(1);
        }
        __builtin_amdgcn_s_cluster_barrier();   // NOP unclustered; fast path if clustered
        __syncthreads();
        __threadfence();        // acquire side (invalidate WGP caches)
    }
}

// ---------------- host launcher ----------------

extern "C" void kernel_launch(void* const* d_in, const int* in_sizes, int n_in,
                              void* d_out, int out_size, void* d_ws, size_t ws_size,
                              hipStream_t stream) {
    const float* x  = (const float*)d_in[0];   // (64,1024,512)
    const float* h0 = (const float*)d_in[1];   // (64,512)
    const float* Wx = (const float*)d_in[2];   // (512,2048)
    const float* Wh = (const float*)d_in[3];   // (512,2048)
    const float* b  = (const float*)d_in[4];   // (2048,)

    char* ws = (char*)d_ws;
    _Float16* pWx  = (_Float16*)(ws);                                  // 2 MB
    _Float16* pWh  = (_Float16*)(ws + (2ull << 20));                   // 2 MB
    _Float16* xh   = (_Float16*)(ws + (4ull << 20));                   // 64 MB
    _Float16* hbuf = (_Float16*)(ws + (68ull << 20));                  // 128 KB (2 bufs)
    int*      ctr  = (int*)     (ws + (68ull << 20) + 131072ull);      // 4 counters

    hipMemsetAsync(ctr, 0, 64, stream);
    lstm_pack_weights<<<8192, 256, 0, stream>>>(Wx, Wh, pWx, pWh);
    lstm_convert_x<<<131072, 256, 0, stream>>>(x, xh);
    lstm_init_h<<<128, 256, 0, stream>>>(h0, hbuf);
    lstm_persistent<<<64, 256, SMEM_TOTAL, stream>>>(xh, pWx, pWh, b, hbuf, ctr, (float*)d_out);
}
